// GraphNet_10075993277153
// MI455X (gfx1250) — matrix operand order repaired and checked by reading, not compiled, not run on hardware
//
#include <hip/hip_runtime.h>

// ---------------------------------------------------------------------------
// GraphNet (MetaLayer) for MI455X / gfx1250.
// All MLP GEMMs run on v_wmma_f32_16x16x32_f16 (f16 in, f32 accumulate).
// One wave (32 lanes, wave32) owns a 16-row tile; activations staged in LDS
// in the WMMA A-operand layout, weights pre-packed into B-operand layout.
// Scatter (segment_sum) via global f32 atomics (L2-resident accumulators).
// ---------------------------------------------------------------------------

#define NN 10000              // nodes
#define NE 100000             // edges
#define NRV (NN * 4)          // node rows  (d2 = 4 subrows)
#define ERV (NE * 4)          // edge rows

typedef __attribute__((ext_vector_type(16))) _Float16 v16h;
typedef __attribute__((ext_vector_type(8)))  _Float16 v8h;
typedef __attribute__((ext_vector_type(8)))  float    v8f;

union V16HU { v16h v; v8h h[2]; };

// A operand (16x32 f16) lane layout:
//   lanes 0-15 : row M=lane, K = c*32 + {0..7, 16..23}
//   lanes16-31 : row M=lane-16, K = c*32 + {8..15, 24..31}
// -> two contiguous 16B LDS reads per lane.
__device__ __forceinline__ v16h load_a16(const _Float16* base, int strideH,
                                         int c, int lane) {
  int r  = lane & 15;
  int hi = (lane >> 4) & 1;
  const _Float16* p = base + r * strideH + c * 32 + hi * 8;
  V16HU u;
  u.h[0] = *(const v8h*)p;
  u.h[1] = *(const v8h*)(p + 16);
  return u.v;
}

// B operand: pre-packed so each lane reads 16 contiguous halves (32B).
__device__ __forceinline__ v16h load_b16(const _Float16* p, int lane) {
  const _Float16* q = p + lane * 16;
  V16HU u;
  u.h[0] = *(const v8h*)q;
  u.h[1] = *(const v8h*)(q + 8);
  return u.v;
}

// out[16][NT*16] = act( A[16][KC*32] @ Wpacked + bias ), f16 result to LDS.
template <int KC, int NT, bool RELU>
__device__ __forceinline__ void gemm_f16(const _Float16* aT, int strideH,
                                         const _Float16* pw, const float* bias,
                                         _Float16* oT, int oStride, int lane) {
#pragma unroll
  for (int t = 0; t < NT; ++t) {
    v8f acc = {};
#pragma unroll
    for (int c = 0; c < KC; ++c) {
      v16h a = load_a16(aT, strideH, c, lane);
      v16h b = load_b16(pw + ((t * KC + c) << 9), lane);
      acc = __builtin_amdgcn_wmma_f32_16x16x32_f16(false, a, false, b,
                                                   (short)0, acc, false, false);
    }
    int   col = t * 16 + (lane & 15);
    float bv  = bias[col];
    int   rb  = (lane >> 4) << 3;  // D layout: vgpr i -> row i + 8*(lane>=16)
#pragma unroll
    for (int i = 0; i < 8; ++i) {
      float v = acc[i] + bv;
      if (RELU) v = fmaxf(v, 0.0f);
      oT[(rb + i) * oStride + col] = (_Float16)v;
    }
  }
}

// --------------------------- utility kernels -------------------------------

__global__ void zero_f32(float* p, long n) {
  long i = (long)blockIdx.x * blockDim.x + threadIdx.x;
  long st = (long)gridDim.x * blockDim.x;
  for (; i < n; i += st) p[i] = 0.0f;
}

// Pack W[K][N] (f32 row-major) into WMMA B lane layout, f16, zero-padded to
// Kp (mult of 32) x Np (mult of 16).
__global__ void pack_w(const float* W, int K, int N, int Kp, int Np,
                       _Float16* dst) {
  int total = Kp * Np;
  int kc    = Kp >> 5;
  for (int idx = blockIdx.x * blockDim.x + threadIdx.x; idx < total;
       idx += gridDim.x * blockDim.x) {
    int j    = idx & 15;
    int lane = (idx >> 4) & 31;
    int blk  = idx >> 9;  // t*kc + c
    int c = blk % kc, t = blk / kc;
    int k = c * 32 + ((j >> 3) << 4) + ((lane >> 4) << 3) + (j & 7);  // lanes>=16 hold K={8..15,24..31}
    int n = t * 16 + (lane & 15);
    float v = (k < K && n < N) ? W[(size_t)k * N + n] : 0.0f;
    dst[idx] = (_Float16)v;
  }
}

// ------------------------------ layer 0 ------------------------------------

// Edge MLP L0: din = 2*32 + 8 + 8(u0=0) = 80 -> Kp 96. Out: new_e (f16) + msg0.
__global__ __launch_bounds__(32) void edge_l0(
    const float* x0, const float* eattr, const int* ei, const float* b0,
    const float* b1, const float* b2, const _Float16* pw0, const _Float16* pw1,
    const _Float16* pw2, _Float16* E1, float* msg0) {
  __shared__ __attribute__((aligned(32))) _Float16 aT[16 * 96];
  __shared__ __attribute__((aligned(32))) _Float16 h1[16 * 64];
  __shared__ __attribute__((aligned(32))) _Float16 h2[16 * 64];
  int lane = threadIdx.x;
  int er0  = blockIdx.x * 16;

  for (int idx = lane; idx < 16 * 96; idx += 32) {
    int i = idx / 96, k = idx - i * 96;
    int er = er0 + i, e = er >> 2, s = er & 3;
    float v = 0.0f;
    if (k < 32) {
      int r = ei[e];
      v = x0[((size_t)r * 4 + s) * 32 + k];
    } else if (k < 64) {
      int c = ei[NE + e];
      v = x0[((size_t)c * 4 + s) * 32 + (k - 32)];
    } else if (k < 72) {
      v = eattr[(size_t)e * 8 + (k - 64)];
    }  // 72..95: u0 = 0 and K-pad
    aT[idx] = (_Float16)v;
  }
  __syncthreads();
  gemm_f16<3, 4, true>(aT, 96, pw0, b0, h1, 64, lane);
  __syncthreads();
  gemm_f16<2, 4, true>(h1, 64, pw1, b1, h2, 64, lane);
  __syncthreads();
#pragma unroll
  for (int t = 0; t < 2; ++t) {
    v8f acc = {};
#pragma unroll
    for (int c = 0; c < 2; ++c) {
      v16h a = load_a16(h2, 64, c, lane);
      v16h b = load_b16(pw2 + ((t * 2 + c) << 9), lane);
      acc = __builtin_amdgcn_wmma_f32_16x16x32_f16(false, a, false, b,
                                                   (short)0, acc, false, false);
    }
    int   col = t * 16 + (lane & 15);
    float bv  = b2[col];
    int   rb  = (lane >> 4) << 3;
#pragma unroll
    for (int i = 0; i < 8; ++i) {
      int er = er0 + rb + i, e = er >> 2, s = er & 3;
      float v = acc[i] + bv;
      E1[(size_t)er * 32 + col] = (_Float16)v;  // new_e, kept f16 for layer 1
      int cn = ei[NE + e];
      atomicAdd(&msg0[((size_t)cn * 4 + s) * 32 + col], v);
    }
  }
}

// Node MLP L0: din = 32 + 32 + 8(u0=0) = 72 -> Kp 96. Out: X1new (f32).
__global__ __launch_bounds__(32) void node_l0(
    const float* x0, const float* msg0, const float* b0, const float* b1,
    const float* b2, const _Float16* pw0, const _Float16* pw1,
    const _Float16* pw2, float* X1) {
  __shared__ __attribute__((aligned(32))) _Float16 aT[16 * 96];
  __shared__ __attribute__((aligned(32))) _Float16 h1[16 * 64];
  __shared__ __attribute__((aligned(32))) _Float16 h2[16 * 64];
  int lane = threadIdx.x;
  int nr0  = blockIdx.x * 16;

  for (int idx = lane; idx < 16 * 96; idx += 32) {
    int i = idx / 96, k = idx - i * 96;
    size_t nr = nr0 + i;
    float  v  = 0.0f;
    if (k < 32)      v = x0[nr * 32 + k];
    else if (k < 64) v = msg0[nr * 32 + (k - 32)];
    aT[idx] = (_Float16)v;
  }
  __syncthreads();
  gemm_f16<3, 4, true>(aT, 96, pw0, b0, h1, 64, lane);
  __syncthreads();
  gemm_f16<2, 4, true>(h1, 64, pw1, b1, h2, 64, lane);
  __syncthreads();
#pragma unroll
  for (int t = 0; t < 2; ++t) {
    v8f acc = {};
#pragma unroll
    for (int c = 0; c < 2; ++c) {
      v16h a = load_a16(h2, 64, c, lane);
      v16h b = load_b16(pw2 + ((t * 2 + c) << 9), lane);
      acc = __builtin_amdgcn_wmma_f32_16x16x32_f16(false, a, false, b,
                                                   (short)0, acc, false, false);
    }
    int   col = t * 16 + (lane & 15);
    float bv  = b2[col];
    int   rb  = (lane >> 4) << 3;
#pragma unroll
    for (int i = 0; i < 8; ++i)
      X1[(size_t)(nr0 + rb + i) * 32 + col] = acc[i] + bv;
  }
}

// Per-graph segment_sum of new_x and msg: register partials + few atomics.
__global__ __launch_bounds__(256) void agg_k(const float* nx, const float* msg,
                                             const int* batch, float* aggN,
                                             float* aggE) {
  int sk = threadIdx.x & 127;   // (s,k) pair
  int sub = threadIdx.x >> 7;   // split node range in two
  int s = sk >> 5, k = sk & 31;
  float an0 = 0, an1 = 0, an2 = 0, an3 = 0;
  float ae0 = 0, ae1 = 0, ae2 = 0, ae3 = 0;
  int per = (NN + gridDim.x - 1) / gridDim.x;
  int n0 = blockIdx.x * per;
  int n1 = n0 + per;
  if (n1 > NN) n1 = NN;
  for (int n = n0 + sub; n < n1; n += 2) {
    int    g   = batch[n];
    size_t off = ((size_t)n * 4 + s) * 32 + k;
    float xv = nx[off], mv = msg[off];
    an0 += (g == 0) ? xv : 0.f; an1 += (g == 1) ? xv : 0.f;
    an2 += (g == 2) ? xv : 0.f; an3 += (g == 3) ? xv : 0.f;
    ae0 += (g == 0) ? mv : 0.f; ae1 += (g == 1) ? mv : 0.f;
    ae2 += (g == 2) ? mv : 0.f; ae3 += (g == 3) ? mv : 0.f;
  }
  int o = s * 32 + k;
  atomicAdd(&aggN[0 * 128 + o], an0); atomicAdd(&aggN[1 * 128 + o], an1);
  atomicAdd(&aggN[2 * 128 + o], an2); atomicAdd(&aggN[3 * 128 + o], an3);
  atomicAdd(&aggE[0 * 128 + o], ae0); atomicAdd(&aggE[1 * 128 + o], ae1);
  atomicAdd(&aggE[2 * 128 + o], ae2); atomicAdd(&aggE[3 * 128 + o], ae3);
}

// Global MLP L0: 16 rows (g,s). din = 32+32+8(u0=0)=72 -> Kp 96 -> 64->64->8.
__global__ __launch_bounds__(32) void glob_l0(
    const float* aggN, const float* aggE, const float* b0, const float* b1,
    const float* b2, const _Float16* pw0, const _Float16* pw1,
    const _Float16* pw2, float* U1) {
  __shared__ __attribute__((aligned(32))) _Float16 aT[16 * 96];
  __shared__ __attribute__((aligned(32))) _Float16 h1[16 * 64];
  __shared__ __attribute__((aligned(32))) _Float16 h2[16 * 64];
  int lane = threadIdx.x;
  for (int idx = lane; idx < 16 * 96; idx += 32) {
    int i = idx / 96, k = idx - i * 96;  // i = g*4 + s
    float v = 0.0f;
    if (k < 32)      v = aggN[i * 32 + k];
    else if (k < 64) v = aggE[i * 32 + (k - 32)];
    aT[idx] = (_Float16)v;
  }
  __syncthreads();
  gemm_f16<3, 4, true>(aT, 96, pw0, b0, h1, 64, lane);
  __syncthreads();
  gemm_f16<2, 4, true>(h1, 64, pw1, b1, h2, 64, lane);
  __syncthreads();
  v8f acc = {};
#pragma unroll
  for (int c = 0; c < 2; ++c) {
    v16h a = load_a16(h2, 64, c, lane);
    v16h b = load_b16(pw2 + (c << 9), lane);
    acc = __builtin_amdgcn_wmma_f32_16x16x32_f16(false, a, false, b, (short)0,
                                                 acc, false, false);
  }
  int col = lane & 15;
  int rb  = (lane >> 4) << 3;
  if (col < 8) {
    float bv = b2[col];
#pragma unroll
    for (int i = 0; i < 8; ++i) U1[(rb + i) * 8 + col] = acc[i] + bv;
  }
}

// ------------------------------ layer 1 ------------------------------------

// Edge MLP L1: din = 2*64 + 40 + 16 = 184 -> Kp 192. Only scatter msg1
// (new_e of the last layer never feeds the output path).
__global__ __launch_bounds__(32) void edge_l1(
    const float* x0, const float* X1, const float* eattr, const _Float16* E1,
    const int* ei, const int* batch, const float* U1, const float* b0,
    const float* b1, const float* b2, const _Float16* pw0, const _Float16* pw1,
    const _Float16* pw2, float* msg1) {
  __shared__ __attribute__((aligned(32))) _Float16 aT[16 * 192];
  __shared__ __attribute__((aligned(32))) _Float16 h1[16 * 64];
  __shared__ __attribute__((aligned(32))) _Float16 h2[16 * 64];
  int lane = threadIdx.x;
  int er0  = blockIdx.x * 16;

  for (int idx = lane; idx < 16 * 192; idx += 32) {
    int i = idx / 192, k = idx - i * 192;
    int er = er0 + i, e = er >> 2, s = er & 3;
    int r = ei[e], cn = ei[NE + e];
    float v = 0.0f;
    if (k < 32)       v = X1[((size_t)r * 4 + s) * 32 + k];          // x1 new
    else if (k < 64)  v = x0[((size_t)r * 4 + s) * 32 + (k - 32)];   // x1 old
    else if (k < 96)  v = X1[((size_t)cn * 4 + s) * 32 + (k - 64)];
    else if (k < 128) v = x0[((size_t)cn * 4 + s) * 32 + (k - 96)];
    else if (k < 160) v = (float)E1[((size_t)e * 4 + s) * 32 + (k - 128)];
    else if (k < 168) v = eattr[(size_t)e * 8 + (k - 160)];
    else if (k < 176) v = U1[(batch[r] * 4 + s) * 8 + (k - 168)];
    // 176..183: u old part = 0; 184..191: K-pad
    aT[idx] = (_Float16)v;
  }
  __syncthreads();
  gemm_f16<6, 4, true>(aT, 192, pw0, b0, h1, 64, lane);
  __syncthreads();
  gemm_f16<2, 4, true>(h1, 64, pw1, b1, h2, 64, lane);
  __syncthreads();
#pragma unroll
  for (int t = 0; t < 2; ++t) {
    v8f acc = {};
#pragma unroll
    for (int c = 0; c < 2; ++c) {
      v16h a = load_a16(h2, 64, c, lane);
      v16h b = load_b16(pw2 + ((t * 2 + c) << 9), lane);
      acc = __builtin_amdgcn_wmma_f32_16x16x32_f16(false, a, false, b,
                                                   (short)0, acc, false, false);
    }
    int   col = t * 16 + (lane & 15);
    float bv  = b2[col];
    int   rb  = (lane >> 4) << 3;
#pragma unroll
    for (int i = 0; i < 8; ++i) {
      int er = er0 + rb + i, e = er >> 2, s = er & 3;
      int cn = ei[NE + e];
      atomicAdd(&msg1[((size_t)cn * 4 + s) * 32 + col], acc[i] + bv);
    }
  }
}

// Node MLP L1: din = 64 + 32 + 16 = 112 -> Kp 128. Out: X2new (f32).
__global__ __launch_bounds__(32) void node_l1(
    const float* x0, const float* X1, const float* msg1, const int* batch,
    const float* U1, const float* b0, const float* b1, const float* b2,
    const _Float16* pw0, const _Float16* pw1, const _Float16* pw2, float* X2) {
  __shared__ __attribute__((aligned(32))) _Float16 aT[16 * 128];
  __shared__ __attribute__((aligned(32))) _Float16 h1[16 * 64];
  __shared__ __attribute__((aligned(32))) _Float16 h2[16 * 64];
  int lane = threadIdx.x;
  int nr0  = blockIdx.x * 16;

  for (int idx = lane; idx < 16 * 128; idx += 32) {
    int i = idx >> 7, k = idx & 127;
    size_t nr = nr0 + i;
    int n = (int)(nr >> 2), s = (int)(nr & 3);
    float v = 0.0f;
    if (k < 32)       v = X1[nr * 32 + k];
    else if (k < 64)  v = x0[nr * 32 + (k - 32)];
    else if (k < 96)  v = msg1[nr * 32 + (k - 64)];
    else if (k < 104) v = U1[(batch[n] * 4 + s) * 8 + (k - 96)];
    aT[idx] = (_Float16)v;
  }
  __syncthreads();
  gemm_f16<4, 4, true>(aT, 128, pw0, b0, h1, 64, lane);
  __syncthreads();
  gemm_f16<2, 4, true>(h1, 64, pw1, b1, h2, 64, lane);
  __syncthreads();
#pragma unroll
  for (int t = 0; t < 2; ++t) {
    v8f acc = {};
#pragma unroll
    for (int c = 0; c < 2; ++c) {
      v16h a = load_a16(h2, 64, c, lane);
      v16h b = load_b16(pw2 + ((t * 2 + c) << 9), lane);
      acc = __builtin_amdgcn_wmma_f32_16x16x32_f16(false, a, false, b,
                                                   (short)0, acc, false, false);
    }
    int   col = t * 16 + (lane & 15);
    float bv  = b2[col];
    int   rb  = (lane >> 4) << 3;
#pragma unroll
    for (int i = 0; i < 8; ++i)
      X2[(size_t)(nr0 + rb + i) * 32 + col] = acc[i] + bv;
  }
}

// Final: out = x2 @ Wout + bout, x2 = [X2new | X1new | x0] (96 dims).
__global__ __launch_bounds__(32) void out_k(const float* x0, const float* X1,
                                            const float* X2, const float* bout,
                                            const _Float16* pwo, float* out) {
  __shared__ __attribute__((aligned(32))) _Float16 aT[16 * 96];
  int lane = threadIdx.x;
  int nr0  = blockIdx.x * 16;

  for (int idx = lane; idx < 16 * 96; idx += 32) {
    int i = idx / 96, k = idx - i * 96;
    size_t nr = nr0 + i;
    float  v;
    if (k < 32)      v = X2[nr * 32 + k];
    else if (k < 64) v = X1[nr * 32 + (k - 32)];
    else             v = x0[nr * 32 + (k - 64)];
    aT[idx] = (_Float16)v;
  }
  __syncthreads();
#pragma unroll
  for (int t = 0; t < 2; ++t) {
    v8f acc = {};
#pragma unroll
    for (int c = 0; c < 3; ++c) {
      v16h a = load_a16(aT, 96, c, lane);
      v16h b = load_b16(pwo + ((t * 3 + c) << 9), lane);
      acc = __builtin_amdgcn_wmma_f32_16x16x32_f16(false, a, false, b,
                                                   (short)0, acc, false, false);
    }
    int   col = t * 16 + (lane & 15);
    float bv  = bout[col];
    int   rb  = (lane >> 4) << 3;
#pragma unroll
    for (int i = 0; i < 8; ++i)
      out[(size_t)(nr0 + rb + i) * 32 + col] = acc[i] + bv;
  }
}

// ------------------------------- host side ---------------------------------

struct MatDesc { int p, K, N; };
// Param leaves in jax pytree order: layers(sorted keys edge<glob<node) then
// node_out; each linear contributes (W, b).
static const MatDesc g_mats[19] = {
    {0, 80, 64},  {2, 64, 64},  {4, 64, 32},   // L0 edge
    {6, 72, 64},  {8, 64, 64},  {10, 64, 8},   // L0 glob
    {12, 72, 64}, {14, 64, 64}, {16, 64, 32},  // L0 node
    {18, 184, 64},{20, 64, 64}, {22, 64, 32},  // L1 edge
    {24, 80, 64}, {26, 64, 64}, {28, 64, 8},   // L1 glob
    {30, 112, 64},{32, 64, 64}, {34, 64, 32},  // L1 node
    {36, 96, 32},                              // node_out
};

extern "C" void kernel_launch(void* const* d_in, const int* in_sizes, int n_in,
                              void* d_out, int out_size, void* d_ws,
                              size_t ws_size, hipStream_t stream) {
  (void)in_sizes; (void)n_in; (void)out_size; (void)ws_size;

  const float* x0    = (const float*)d_in[0];
  const float* eattr = (const float*)d_in[1];
  const float* P[38];
  for (int i = 0; i < 38; ++i) P[i] = (const float*)d_in[2 + i];
  const int* ei    = (const int*)d_in[40];
  const int* batch = (const int*)d_in[41];
  float*     out   = (float*)d_out;

  // ---- workspace carve-up (deterministic, no allocations) ----
  char*  w   = (char*)d_ws;
  size_t off = 0;
  auto alloc = [&](size_t bytes) {
    void* p = w + off;
    off = (off + bytes + 255) & ~(size_t)255;
    return p;
  };

  const size_t msgF = (size_t)NRV * 32;           // floats per msg buffer
  float* zbase = (float*)alloc((2 * msgF + 1024) * 4);  // zeroed accumulators
  float* msg0  = zbase;
  float* msg1  = zbase + msgF;
  float* aggN0 = msg1 + msgF;
  float* aggE0 = aggN0 + 512;

  _Float16* E1 = (_Float16*)alloc((size_t)ERV * 32 * 2);  // new_e L0, f16
  float*    X1 = (float*)alloc((size_t)NRV * 32 * 4);     // new_x L0
  float*    X2 = (float*)alloc((size_t)NRV * 32 * 4);     // new_x L1
  float*    U1 = (float*)alloc(16 * 8 * 4);               // new_u L0

  _Float16* pw[19];
  for (int i = 0; i < 19; ++i) {
    int Kp = (g_mats[i].K + 31) & ~31;
    int Np = (g_mats[i].N + 15) & ~15;
    pw[i]  = (_Float16*)alloc((size_t)Kp * Np * 2);
    int total = Kp * Np;
    pack_w<<<(total + 255) / 256, 256, 0, stream>>>(P[g_mats[i].p],
                                                    g_mats[i].K, g_mats[i].N,
                                                    Kp, Np, pw[i]);
  }

  zero_f32<<<512, 256, 0, stream>>>(zbase, (long)(2 * msgF + 1024));

  // ---- layer 0 ----
  edge_l0<<<ERV / 16, 32, 0, stream>>>(x0, eattr, ei, P[1], P[3], P[5],
                                       pw[0], pw[1], pw[2], E1, msg0);
  node_l0<<<NRV / 16, 32, 0, stream>>>(x0, msg0, P[13], P[15], P[17],
                                       pw[6], pw[7], pw[8], X1);
  agg_k<<<50, 256, 0, stream>>>(X1, msg0, batch, aggN0, aggE0);
  glob_l0<<<1, 32, 0, stream>>>(aggN0, aggE0, P[7], P[9], P[11],
                                pw[3], pw[4], pw[5], U1);

  // ---- layer 1 (u2 / agg of layer 1 never reach the output; skipped) ----
  edge_l1<<<ERV / 16, 32, 0, stream>>>(x0, X1, eattr, E1, ei, batch, U1,
                                       P[19], P[21], P[23], pw[9], pw[10],
                                       pw[11], msg1);
  node_l1<<<NRV / 16, 32, 0, stream>>>(x0, X1, msg1, batch, U1, P[31], P[33],
                                       P[35], pw[15], pw[16], pw[17], X2);

  // ---- output projection ----
  out_k<<<NRV / 16, 32, 0, stream>>>(x0, X1, X2, P[37], pw[18], out);
}
